// Arch22GraphEncoder_3229815406967
// MI455X (gfx1250) — matrix operand order repaired and checked by reading, not compiled, not run on hardware
//
#include <hip/hip_runtime.h>
#include <hip/hip_bf16.h>
#include <math.h>

// ---------------- problem constants ----------------
#define B_G   128
#define M_P   64
#define N_PG  64
#define KSEQ  9
#define H     256
#define S_TOT (B_G * M_P)          // 8192 patterns
#define ROWS  (S_TOT * KSEQ)       // 73728 token rows
#define E_TOT (B_G * N_PG * 4)     // 32768 edges
#define PE_STEPS 8
#define FEAT_PAD 64                // 16 + 8 + 16 = 40, padded to 64 for WMMA K

typedef __attribute__((ext_vector_type(16))) _Float16 v16h;
typedef __attribute__((ext_vector_type(8)))  float    v8f;

union Frag { v16h h; unsigned u[8]; };

// =====================================================================
// Generic WMMA GEMM:  C[M x N] = A[M x K](f16, row-major) * W[N x K]^T (f16)
// + bias (f32, per column). One 16x16 D-tile per wave, 8 waves -> 64x32 block.
// M % 64 == 0, N % 32 == 0, K % 32 == 0 for every call below.
// =====================================================================
__global__ __launch_bounds__(256) void gemm_wmma_f16(
    const _Float16* __restrict__ A, const _Float16* __restrict__ W,
    const float* __restrict__ bias, float* __restrict__ Cf,
    _Float16* __restrict__ Ch, int M, int N, int Kd, int relu)
{
    const int wave = threadIdx.x >> 5;
    const int lane = threadIdx.x & 31;
    const int bm   = blockIdx.x * 64 + (wave & 3) * 16;
    const int bn   = blockIdx.y * 32 + (wave >> 2) * 16;
    const int mrow = lane & 15;        // A row / W row (n) within tile
    const int hi   = lane >> 4;        // lane-half selector

    v8f acc = {};
    const _Float16* Arow = A + (size_t)(bm + mrow) * Kd;
    const _Float16* Wrow = W + (size_t)(bn + mrow) * Kd + hi * 16;

    for (int k0 = 0; k0 < Kd; k0 += 32) {
        Frag a, b;
        // A fragment: 16x32 f16: VGPR v<4 -> K=2v,2v+1 (+16 for hi lanes)
        //             VGPR v>=4 -> K=8+2(v-4),... (+16 for hi lanes)
#pragma unroll
        for (int v = 0; v < 4; ++v)
            a.u[v] = *(const unsigned*)(Arow + k0 + 2 * v + hi * 16);
#pragma unroll
        for (int v = 0; v < 4; ++v)
            a.u[4 + v] = *(const unsigned*)(Arow + k0 + 8 + 2 * v + hi * 16);
        // B fragment: 32x16 f16, lane n = lane%16, K range [hi*16, hi*16+16)
#pragma unroll
        for (int v = 0; v < 8; ++v)
            b.u[v] = *(const unsigned*)(Wrow + k0 + 2 * v);

        acc = __builtin_amdgcn_wmma_f32_16x16x32_f16(
            false, a.h, false, b.h, (short)0, acc, false, false);
    }

    const int   n  = bn + mrow;
    const float bv = bias ? bias[n] : 0.0f;
#pragma unroll
    for (int j = 0; j < 8; ++j) {
        const int m = bm + j + hi * 8;       // C layout: VGPR j -> row j (+8 hi)
        float v = acc[j] + bv;
        if (relu) v = fmaxf(v, 0.0f);
        if (Cf) Cf[(size_t)m * N + n] = v;
        if (Ch) Ch[(size_t)m * N + n] = (_Float16)v;
    }
}

// =====================================================================
// Small utility kernels
// =====================================================================
__global__ void zero_f32_kernel(float* p, int n) {
    int i = blockIdx.x * 256 + threadIdx.x;
    if (i < n) p[i] = 0.0f;
}
__global__ void zero_f16_kernel(_Float16* p, int n) {
    int i = blockIdx.x * 256 + threadIdx.x;
    if (i < n) p[i] = (_Float16)0.0f;
}
// f32 -> f16 with optional column padding (src cols sc -> dst cols dc, zero pad)
__global__ void cvt_pad_f16_kernel(const float* __restrict__ src,
                                   _Float16* __restrict__ dst,
                                   int rows, int sc, int dc) {
    int i = blockIdx.x * 256 + threadIdx.x;
    int total = rows * dc;
    if (i >= total) return;
    int r = i / dc, c = i - r * dc;
    dst[i] = (c < sc) ? (_Float16)src[(size_t)r * sc + c] : (_Float16)0.0f;
}

// =====================================================================
// RWSE: per graph build A (64x64) in LDS, P = A / max(rowsum,1),
// record diag(P^t) for t = 1..8.  One block per graph.
// =====================================================================
__global__ __launch_bounds__(256) void rwse_kernel(
    const int* __restrict__ esrc, const int* __restrict__ edst,
    float* __restrict__ rwse /* (B_G*N_PG) x 8 */)
{
    __shared__ float Ab[64 * 64];
    __shared__ float Mb[64 * 64];
    __shared__ float Tb[64 * 64];
    __shared__ float rs[64];
    const int g = blockIdx.x, tid = threadIdx.x;

    for (int i = tid; i < 4096; i += 256) Ab[i] = 0.0f;
    __syncthreads();
    for (int e = tid; e < E_TOT; e += 256) {
        int s = esrc[e];
        if ((s >> 6) == g) atomicAdd(&Ab[(s & 63) * 64 + (edst[e] & 63)], 1.0f);
    }
    __syncthreads();
    if (tid < 64) {
        float sum = 0.0f;
        for (int j = 0; j < 64; ++j) sum += Ab[tid * 64 + j];
        rs[tid] = fmaxf(sum, 1.0f);
    }
    __syncthreads();
    for (int i = tid; i < 4096; i += 256) {
        float p = Ab[i] / rs[i >> 6];
        Ab[i] = p;        // P
        Mb[i] = p;        // M = P
    }
    __syncthreads();
    for (int t = 0; t < PE_STEPS; ++t) {
        if (tid < 64) rwse[(size_t)(g * 64 + tid) * PE_STEPS + t] = Mb[tid * 64 + tid];
        __syncthreads();
        for (int i = tid; i < 4096; i += 256) {
            int r = i >> 6, c = i & 63;
            float acc = 0.0f;
            for (int k = 0; k < 64; ++k) acc += Mb[r * 64 + k] * Ab[k * 64 + c];
            Tb[i] = acc;
        }
        __syncthreads();
        for (int i = tid; i < 4096; i += 256) Mb[i] = Tb[i];
        __syncthreads();
    }
}

// =====================================================================
// Build padded token features in "gpm order" (row = (m*B_G + b)*K + k):
// [atom_emb(16) | rwse(8) | bond(16) | zero pad(24)] as f16, 64 cols.
// =====================================================================
__global__ void build_feat_kernel(
    const int* __restrict__ x, const int* __restrict__ edge_attr,
    const int* __restrict__ nodes_sampled, const int* __restrict__ edge_src_global,
    const float* __restrict__ atom_emb, const float* __restrict__ bond_emb,
    const float* __restrict__ rwse, _Float16* __restrict__ feat)
{
    int r = blockIdx.x * 256 + threadIdx.x;
    if (r >= ROWS) return;
    int k  = r % KSEQ;
    int sp = r / KSEQ;                 // sp = m*B_G + b  (gpm order)
    int m  = sp / B_G, b = sp - m * B_G;
    int so = b * M_P + m;              // original pattern row
    int node = nodes_sampled[so * KSEQ + k];
    int ns = node > 0 ? node : 0;
    _Float16* out = feat + (size_t)r * FEAT_PAD;
    const float* ae = atom_emb + x[ns] * 16;
#pragma unroll
    for (int c = 0; c < 16; ++c) out[c] = (_Float16)ae[c];
    const float* rw = rwse + (size_t)ns * PE_STEPS;
#pragma unroll
    for (int c = 0; c < 8; ++c) out[16 + c] = (_Float16)rw[c];
    if (k == 0) {
#pragma unroll
        for (int c = 0; c < 16; ++c) out[24 + c] = (_Float16)0.0f;
    } else {
        int e  = edge_src_global[so * (KSEQ - 1) + (k - 1)];
        const float* be = bond_emb + (edge_attr[e] - 1) * 16;
#pragma unroll
        for (int c = 0; c < 16; ++c) out[24 + c] = (_Float16)be[c];
    }
#pragma unroll
    for (int c = 40; c < FEAT_PAD; ++c) out[c] = (_Float16)0.0f;
}

// =====================================================================
// GRU step t: gx from 9-wide adjacency row (on the fly) + gh (from WMMA GEMM),
// gate math, update h / h16, accumulate pe_sum.  One block per pattern.
// =====================================================================
__global__ __launch_bounds__(256) void gru_step_kernel(
    const int* __restrict__ nodes_sampled, const float* __restrict__ w_ih,
    const float* __restrict__ b_ih, const float* __restrict__ gh,
    float* __restrict__ h, _Float16* __restrict__ h16,
    float* __restrict__ pe_sum, int t)
{
    const int sp = blockIdx.x;         // gpm order
    const int j  = threadIdx.x;        // 0..255
    __shared__ int p[KSEQ];
    if (j < KSEQ) {
        int m = sp / B_G, b = sp - m * B_G;
        int v = nodes_sampled[(b * M_P + m) * KSEQ + j];
        p[j] = v > 0 ? v : 0;
    }
    __syncthreads();
    const int pt = p[t];
    float gxr = b_ih[j], gxz = b_ih[H + j], gxn = b_ih[2 * H + j];
#pragma unroll
    for (int k = 0; k < KSEQ; ++k) {
        float a = (p[k] == pt) ? 1.0f : 0.0f;
        gxr += a * w_ih[(size_t)j * KSEQ + k];
        gxz += a * w_ih[(size_t)(H + j) * KSEQ + k];
        gxn += a * w_ih[(size_t)(2 * H + j) * KSEQ + k];
    }
    const float* g = gh + (size_t)sp * 3 * H;
    float r = 1.0f / (1.0f + expf(-(gxr + g[j])));
    float z = 1.0f / (1.0f + expf(-(gxz + g[H + j])));
    float n = tanhf(gxn + r * g[2 * H + j]);
    size_t idx = (size_t)sp * H + j;
    float hn = (1.0f - z) * n + z * h[idx];
    h[idx] = hn;
    h16[idx] = (_Float16)hn;
    pe_sum[idx] += hn;
}

__global__ void pe_mean_kernel(const float* __restrict__ pe_sum,
                               _Float16* __restrict__ pe16, int n) {
    int i = blockIdx.x * 256 + threadIdx.x;
    if (i < n) pe16[i] = (_Float16)(pe_sum[i] * (1.0f / 9.0f));
}

// =====================================================================
// Intra attention: 1 head, dh=256, seq len 9, multiscale key mask.
// One block per pattern.
// =====================================================================
__global__ __launch_bounds__(256) void attn_intra_kernel(
    const _Float16* __restrict__ qkv /* ROWS x 768 */,
    _Float16* __restrict__ out /* ROWS x 256 */)
{
    const int s = blockIdx.x, tid = threadIdx.x;
    const int m = s / B_G;
    const int L = 2 * (m / 16) + 3;    // valid key count per multiscale
    __shared__ _Float16 q[KSEQ * H], k[KSEQ * H], v[KSEQ * H];
    __shared__ float aw[KSEQ * KSEQ];
    const _Float16* base = qkv + (size_t)s * KSEQ * 3 * H;
    for (int i = tid; i < KSEQ * H; i += 256) {
        int row = i / H, c = i - row * H;
        q[i] = base[row * 3 * H + c];
        k[i] = base[row * 3 * H + H + c];
        v[i] = base[row * 3 * H + 2 * H + c];
    }
    __syncthreads();
    if (tid < KSEQ * KSEQ) {
        int i = tid / KSEQ, j = tid - i * KSEQ;
        float acc = 0.0f;
        for (int c = 0; c < H; ++c) acc += (float)q[i * H + c] * (float)k[j * H + c];
        aw[tid] = acc * (1.0f / 16.0f);   // 1/sqrt(256)
    }
    __syncthreads();
    if (tid < KSEQ) {
        float mx = -1e30f;
        for (int j = 0; j < L; ++j) mx = fmaxf(mx, aw[tid * KSEQ + j]);
        float sum = 0.0f;
        for (int j = 0; j < KSEQ; ++j) {
            float e = (j < L) ? expf(aw[tid * KSEQ + j] - mx) : 0.0f;
            aw[tid * KSEQ + j] = e;
            sum += e;
        }
        float inv = 1.0f / sum;
        for (int j = 0; j < KSEQ; ++j) aw[tid * KSEQ + j] *= inv;
    }
    __syncthreads();
    for (int o = tid; o < KSEQ * H; o += 256) {
        int i = o / H, c = o - i * H;
        float acc = 0.0f;
#pragma unroll
        for (int j = 0; j < KSEQ; ++j) acc += aw[i * KSEQ + j] * (float)v[j * H + c];
        out[((size_t)s * KSEQ + i) * H + c] = (_Float16)acc;
    }
}

// =====================================================================
// Inter attention: 64 batches x 4 heads, L=128, dh=64. Block = (batch,head),
// thread = query position (two-pass softmax, all in regs + LDS K/V tiles).
// =====================================================================
__global__ __launch_bounds__(128) void attn_inter_kernel(
    const _Float16* __restrict__ qkv /* 8192 x 768 */,
    _Float16* __restrict__ out /* 8192 x 256 */)
{
    const int mb = blockIdx.x >> 2;
    const int hh = blockIdx.x & 3;
    const int tid = threadIdx.x;       // query b in [0,128)
    __shared__ _Float16 Ks[128 * 64], Vs[128 * 64];
    const _Float16* base = qkv + (size_t)mb * 128 * 3 * H;
    for (int i = tid; i < 128 * 64; i += 128) {
        int row = i >> 6, c = i & 63;
        Ks[i] = base[row * 3 * H + H + hh * 64 + c];
        Vs[i] = base[row * 3 * H + 2 * H + hh * 64 + c];
    }
    __syncthreads();
    float q[64];
    const _Float16* qp = base + (size_t)tid * 3 * H + hh * 64;
#pragma unroll
    for (int c = 0; c < 64; ++c) q[c] = (float)qp[c];
    float mx = -1e30f;
    for (int j = 0; j < 128; ++j) {
        float acc = 0.0f;
#pragma unroll
        for (int c = 0; c < 64; ++c) acc += q[c] * (float)Ks[j * 64 + c];
        mx = fmaxf(mx, acc * 0.125f);  // 1/sqrt(64)
    }
    float o[64];
#pragma unroll
    for (int c = 0; c < 64; ++c) o[c] = 0.0f;
    float l = 0.0f;
    for (int j = 0; j < 128; ++j) {
        float acc = 0.0f;
#pragma unroll
        for (int c = 0; c < 64; ++c) acc += q[c] * (float)Ks[j * 64 + c];
        float p = expf(acc * 0.125f - mx);
        l += p;
#pragma unroll
        for (int c = 0; c < 64; ++c) o[c] += p * (float)Vs[j * 64 + c];
    }
    float inv = 1.0f / l;
    _Float16* op = out + (size_t)(mb * 128 + tid) * H + hh * 64;
#pragma unroll
    for (int c = 0; c < 64; ++c) op[c] = (_Float16)(o[c] * inv);
}

// =====================================================================
// y = LayerNorm(x (+ r)) * w + b ; stores f32 and/or f16. One block per row.
// =====================================================================
__global__ __launch_bounds__(256) void res_ln_kernel(
    const float* __restrict__ x, const float* __restrict__ r,
    const float* __restrict__ w, const float* __restrict__ b,
    float* __restrict__ yf, _Float16* __restrict__ yh)
{
    const int row = blockIdx.x, tid = threadIdx.x;
    __shared__ float red[256];
    float v = x[(size_t)row * H + tid];
    if (r) v += r[(size_t)row * H + tid];
    red[tid] = v; __syncthreads();
    for (int s = 128; s > 0; s >>= 1) { if (tid < s) red[tid] += red[tid + s]; __syncthreads(); }
    float mu = red[0] * (1.0f / H);
    __syncthreads();
    float d = v - mu;
    red[tid] = d * d; __syncthreads();
    for (int s = 128; s > 0; s >>= 1) { if (tid < s) red[tid] += red[tid + s]; __syncthreads(); }
    float var = red[0] * (1.0f / H);
    float y = d * rsqrtf(var + 1e-5f) * w[tid] + b[tid];
    if (yf) yf[(size_t)row * H + tid] = y;
    if (yh) yh[(size_t)row * H + tid] = (_Float16)y;
}

// pf = mean_k(h_enc) + pe_vec
__global__ void make_pf_kernel(const float* __restrict__ henc,
                               const float* __restrict__ pe_vec,
                               float* __restrict__ pf)
{
    int s = blockIdx.x, c = threadIdx.x;
    float acc = 0.0f;
#pragma unroll
    for (int k = 0; k < KSEQ; ++k) acc += henc[((size_t)s * KSEQ + k) * H + c];
    size_t i = (size_t)s * H + c;
    pf[i] = acc * (1.0f / 9.0f) + pe_vec[i];
}

// out[b,c] = mean over m of (pf + enc_out)[m*B_G + b, c]
__global__ void final_out_kernel(const float* __restrict__ pf,
                                 const float* __restrict__ encout,
                                 float* __restrict__ out)
{
    int b = blockIdx.x, c = threadIdx.x;
    float acc = 0.0f;
    for (int m = 0; m < M_P; ++m) {
        size_t r = ((size_t)m * B_G + b) * H + c;
        acc += pf[r] + encout[r];
    }
    out[(size_t)b * H + c] = acc * (1.0f / M_P);
}

// =====================================================================
// Host-side launch
// =====================================================================
static inline char* carve(char*& p, size_t bytes) {
    char* r = p;
    p += (bytes + 255) & ~(size_t)255;
    return r;
}

extern "C" void kernel_launch(void* const* d_in, const int* in_sizes, int n_in,
                              void* d_out, int out_size, void* d_ws, size_t ws_size,
                              hipStream_t stream) {
    (void)in_sizes; (void)n_in; (void)out_size; (void)ws_size;
    const int* x_in          = (const int*)d_in[0];
    const int* edge_attr     = (const int*)d_in[1];
    const int* edge_index    = (const int*)d_in[2];
    const int* nodes_sampled = (const int*)d_in[3];
    const int* edge_src_glob = (const int*)d_in[4];
    const float* atom_emb = (const float*)d_in[7];
    const float* bond_emb = (const float*)d_in[8];
    const float* ip_w  = (const float*)d_in[9];
    const float* ip_b  = (const float*)d_in[10];
    const float* g_wih = (const float*)d_in[11];
    const float* g_whh = (const float*)d_in[12];
    const float* g_bih = (const float*)d_in[13];
    const float* g_bhh = (const float*)d_in[14];
    const float* pep_w = (const float*)d_in[15];
    const float* pep_b = (const float*)d_in[16];
    // intra encoder params (17..28)
    const float* A_inw = (const float*)d_in[17];
    const float* A_inb = (const float*)d_in[18];
    const float* A_ow  = (const float*)d_in[19];
    const float* A_ob  = (const float*)d_in[20];
    const float* A_l1w = (const float*)d_in[21];
    const float* A_l1b = (const float*)d_in[22];
    const float* A_l2w = (const float*)d_in[23];
    const float* A_l2b = (const float*)d_in[24];
    const float* A_f1w = (const float*)d_in[25];
    const float* A_f1b = (const float*)d_in[26];
    const float* A_f2w = (const float*)d_in[27];
    const float* A_f2b = (const float*)d_in[28];
    const float* nrm_w = (const float*)d_in[29];
    const float* nrm_b = (const float*)d_in[30];
    // inter encoder params (31..42)
    const float* B_inw = (const float*)d_in[31];
    const float* B_inb = (const float*)d_in[32];
    const float* B_ow  = (const float*)d_in[33];
    const float* B_ob  = (const float*)d_in[34];
    const float* B_l1w = (const float*)d_in[35];
    const float* B_l1b = (const float*)d_in[36];
    const float* B_l2w = (const float*)d_in[37];
    const float* B_l2b = (const float*)d_in[38];
    const float* B_f1w = (const float*)d_in[39];
    const float* B_f1b = (const float*)d_in[40];
    const float* B_f2w = (const float*)d_in[41];
    const float* B_f2b = (const float*)d_in[42];

    // ---------------- workspace carve ----------------
    char* wp = (char*)d_ws;
    float*     rwse    = (float*)    carve(wp, (size_t)S_TOT * PE_STEPS * 4);
    // big f16 region R: feat16 -> qkv16 -> ffn16 (sequential lifetimes)
    char*      Rreg    =             carve(wp, (size_t)ROWS * 1024 * 2);
    _Float16*  feat16  = (_Float16*)Rreg;
    _Float16*  qkv16   = (_Float16*)Rreg;
    _Float16*  ffn16   = (_Float16*)Rreg;
    float*     W0      = (float*)    carve(wp, (size_t)ROWS * H * 4);   // h_gpm / x1
    _Float16*  W1      = (_Float16*)carve(wp, (size_t)ROWS * H * 2);    // h_gpm16 / x1_16
    _Float16*  attn16  = (_Float16*)carve(wp, (size_t)ROWS * H * 2);
    float*     W4      = (float*)    carve(wp, (size_t)ROWS * H * 4);   // mha / ffn2 / h_enc
    // f16 weights
    _Float16* ipw16  = (_Float16*)carve(wp, (size_t)H * FEAT_PAD * 2);
    _Float16* whh16  = (_Float16*)carve(wp, (size_t)3 * H * H * 2);
    _Float16* pepw16 = (_Float16*)carve(wp, (size_t)H * H * 2);
    _Float16* Ainw16 = (_Float16*)carve(wp, (size_t)3 * H * H * 2);
    _Float16* Aow16  = (_Float16*)carve(wp, (size_t)H * H * 2);
    _Float16* Af1w16 = (_Float16*)carve(wp, (size_t)4 * H * H * 2);
    _Float16* Af2w16 = (_Float16*)carve(wp, (size_t)H * 4 * H * 2);
    _Float16* Binw16 = (_Float16*)carve(wp, (size_t)3 * H * H * 2);
    _Float16* Bow16  = (_Float16*)carve(wp, (size_t)H * H * 2);
    _Float16* Bf1w16 = (_Float16*)carve(wp, (size_t)4 * H * H * 2);
    _Float16* Bf2w16 = (_Float16*)carve(wp, (size_t)H * 4 * H * 2);
    // GRU state
    float*    hbuf   = (float*)    carve(wp, (size_t)S_TOT * H * 4);
    _Float16* h16    = (_Float16*)carve(wp, (size_t)S_TOT * H * 2);
    float*    ghbuf  = (float*)    carve(wp, (size_t)S_TOT * 3 * H * 4);
    float*    pe_sum = (float*)    carve(wp, (size_t)S_TOT * H * 4);
    _Float16* pe16   = (_Float16*)carve(wp, (size_t)S_TOT * H * 2);
    float*    pe_vec = (float*)    carve(wp, (size_t)S_TOT * H * 4);
    // inter stage
    float*    pf      = (float*)    carve(wp, (size_t)S_TOT * H * 4);
    float*    ln_in   = (float*)    carve(wp, (size_t)S_TOT * H * 4);
    _Float16* ln_in16 = (_Float16*)carve(wp, (size_t)S_TOT * H * 2);
    _Float16* qkv2    = (_Float16*)carve(wp, (size_t)S_TOT * 3 * H * 2);
    _Float16* attn2   = (_Float16*)carve(wp, (size_t)S_TOT * H * 2);
    float*    mha2    = (float*)    carve(wp, (size_t)S_TOT * H * 4);
    float*    y1      = (float*)    carve(wp, (size_t)S_TOT * H * 4);
    _Float16* y1_16   = (_Float16*)carve(wp, (size_t)S_TOT * H * 2);
    _Float16* ffn2_16 = (_Float16*)carve(wp, (size_t)S_TOT * 4 * H * 2);
    float*    f2out   = (float*)    carve(wp, (size_t)S_TOT * H * 4);
    float*    encout  = (float*)    carve(wp, (size_t)S_TOT * H * 4);

    auto cvt = [&](const float* s, _Float16* d, int rows, int sc, int dc) {
        int tot = rows * dc;
        cvt_pad_f16_kernel<<<(tot + 255) / 256, 256, 0, stream>>>(s, d, rows, sc, dc);
    };

    // ---- weight conversion ----
    cvt(ip_w,  ipw16,  H, 40, FEAT_PAD);
    cvt(g_whh, whh16,  3 * H, H, H);
    cvt(pep_w, pepw16, H, H, H);
    cvt(A_inw, Ainw16, 3 * H, H, H);
    cvt(A_ow,  Aow16,  H, H, H);
    cvt(A_f1w, Af1w16, 4 * H, H, H);
    cvt(A_f2w, Af2w16, H, 4 * H, 4 * H);
    cvt(B_inw, Binw16, 3 * H, H, H);
    cvt(B_ow,  Bow16,  H, H, H);
    cvt(B_f1w, Bf1w16, 4 * H, H, H);
    cvt(B_f2w, Bf2w16, H, 4 * H, 4 * H);

    // ---- RWSE ----
    rwse_kernel<<<B_G, 256, 0, stream>>>(edge_index, edge_index + E_TOT, rwse);

    // ---- features + input projection (WMMA) -> h_gpm f32 in W0 ----
    build_feat_kernel<<<(ROWS + 255) / 256, 256, 0, stream>>>(
        x_in, edge_attr, nodes_sampled, edge_src_glob, atom_emb, bond_emb, rwse, feat16);
    gemm_wmma_f16<<<dim3(ROWS / 64, H / 32), 256, 0, stream>>>(
        feat16, ipw16, ip_b, W0, (_Float16*)nullptr, ROWS, H, FEAT_PAD, 0);

    // ---- GRU over 9 steps (gh via WMMA each step) ----
    {
        int n = S_TOT * H;
        zero_f32_kernel<<<(n + 255) / 256, 256, 0, stream>>>(hbuf, n);
        zero_f32_kernel<<<(n + 255) / 256, 256, 0, stream>>>(pe_sum, n);
        zero_f16_kernel<<<(n + 255) / 256, 256, 0, stream>>>(h16, n);
    }
    for (int t = 0; t < KSEQ; ++t) {
        gemm_wmma_f16<<<dim3(S_TOT / 64, (3 * H) / 32), 256, 0, stream>>>(
            h16, whh16, g_bhh, ghbuf, (_Float16*)nullptr, S_TOT, 3 * H, H, 0);
        gru_step_kernel<<<S_TOT, 256, 0, stream>>>(
            nodes_sampled, g_wih, g_bih, ghbuf, hbuf, h16, pe_sum, t);
    }
    pe_mean_kernel<<<(S_TOT * H + 255) / 256, 256, 0, stream>>>(pe_sum, pe16, S_TOT * H);
    gemm_wmma_f16<<<dim3(S_TOT / 64, H / 32), 256, 0, stream>>>(
        pe16, pepw16, pep_b, pe_vec, (_Float16*)nullptr, S_TOT, H, H, 0);

    // ---- intra encoder layer (seq len 9, 1 head) ----
    cvt(W0, W1, ROWS, H, H);                                   // h_gpm -> f16
    gemm_wmma_f16<<<dim3(ROWS / 64, (3 * H) / 32), 256, 0, stream>>>(
        W1, Ainw16, A_inb, (float*)nullptr, qkv16, ROWS, 3 * H, H, 0);
    attn_intra_kernel<<<S_TOT, 256, 0, stream>>>(qkv16, attn16);
    gemm_wmma_f16<<<dim3(ROWS / 64, H / 32), 256, 0, stream>>>(
        attn16, Aow16, A_ob, W4, (_Float16*)nullptr, ROWS, H, H, 0);
    res_ln_kernel<<<ROWS, 256, 0, stream>>>(W0, W4, A_l1w, A_l1b, W0, W1);  // x1
    gemm_wmma_f16<<<dim3(ROWS / 64, (4 * H) / 32), 256, 0, stream>>>(
        W1, Af1w16, A_f1b, (float*)nullptr, ffn16, ROWS, 4 * H, H, 1);      // ReLU
    gemm_wmma_f16<<<dim3(ROWS / 64, H / 32), 256, 0, stream>>>(
        ffn16, Af2w16, A_f2b, W4, (_Float16*)nullptr, ROWS, H, 4 * H, 0);
    res_ln_kernel<<<ROWS, 256, 0, stream>>>(W0, W4, A_l2w, A_l2b, W4, (_Float16*)nullptr); // h_enc

    // ---- pf = mean_k(h_enc) + pe_vec ----
    make_pf_kernel<<<S_TOT, 256, 0, stream>>>(W4, pe_vec, pf);

    // ---- inter encoder layer (batch 64, seq 128, 4 heads) ----
    res_ln_kernel<<<S_TOT, 256, 0, stream>>>(pf, (const float*)nullptr, nrm_w, nrm_b, ln_in, ln_in16);
    gemm_wmma_f16<<<dim3(S_TOT / 64, (3 * H) / 32), 256, 0, stream>>>(
        ln_in16, Binw16, B_inb, (float*)nullptr, qkv2, S_TOT, 3 * H, H, 0);
    attn_inter_kernel<<<M_P * 4, 128, 0, stream>>>(qkv2, attn2);
    gemm_wmma_f16<<<dim3(S_TOT / 64, H / 32), 256, 0, stream>>>(
        attn2, Bow16, B_ob, mha2, (_Float16*)nullptr, S_TOT, H, H, 0);
    res_ln_kernel<<<S_TOT, 256, 0, stream>>>(ln_in, mha2, B_l1w, B_l1b, y1, y1_16);
    gemm_wmma_f16<<<dim3(S_TOT / 64, (4 * H) / 32), 256, 0, stream>>>(
        y1_16, Bf1w16, B_f1b, (float*)nullptr, ffn2_16, S_TOT, 4 * H, H, 1);
    gemm_wmma_f16<<<dim3(S_TOT / 64, H / 32), 256, 0, stream>>>(
        ffn2_16, Bf2w16, B_f2b, f2out, (_Float16*)nullptr, S_TOT, H, 4 * H, 0);
    res_ln_kernel<<<S_TOT, 256, 0, stream>>>(y1, f2out, B_l2w, B_l2b, encout, (_Float16*)nullptr);

    // ---- final: mean over patterns ----
    final_out_kernel<<<B_G, 256, 0, stream>>>(pf, encout, (float*)d_out);
}
